// HyperFC_15839839387906
// MI455X (gfx1250) — matrix operand order repaired and checked by reading, not compiled, not run on hardware
//
#include <hip/hip_runtime.h>
#include <math.h>

// ---------------------------------------------------------------------------
// HyperFC for MI455X (gfx1250, wave32).
//   A) layernorm + gelu(ctx@W1^T+b1)            -> h      (f16, ws)
//   B) params = alpha*(h@W2^T)+slow  via WMMA   -> params (f16, ws)
//   C) out[b,u,v,o] = sum_i params[u,o,i]*in[b,u,v,i] + params[u,NW+o]
//      via WMMA, with double-buffered GLOBAL_LOAD_ASYNC_TO_LDS_B128 staging.
// f16 operands / f32 accumulation using v_wmma_f32_16x16x32_f16.
// ---------------------------------------------------------------------------

typedef __attribute__((ext_vector_type(16))) _Float16 v16h;
typedef __attribute__((ext_vector_type(8)))  _Float16 v8h;
typedef __attribute__((ext_vector_type(8)))  float    v8f;

#define I_DIM   256
#define O_DIM   256
#define C_DIM   256
#define CAP_DIM 256
#define U_DIM   512
#define B_DIM   8
#define V_DIM   64
#define NW      (O_DIM * I_DIM)   // 65536
#define NP      (NW + O_DIM)      // 65792
#define EPS_LN  1e-5f

// Async copy of 16 bytes global -> LDS (CDNA5, ASYNCcnt-tracked), GV addressing.
__device__ __forceinline__ void async_copy_b128(unsigned lds_off, const void* gptr) {
    asm volatile("global_load_async_to_lds_b128 %0, %1, off"
                 :: "v"(lds_off), "v"(gptr)
                 : "memory");
}

// --------------------------- Kernel A: h = gelu(LN(ctx)@W1^T + b1) ----------
__global__ __launch_bounds__(256) void hyperfc_h_kernel(
    const float* __restrict__ context, const float* __restrict__ gamma,
    const float* __restrict__ beta,    const float* __restrict__ W1,
    const float* __restrict__ b1,      _Float16* __restrict__ h_out)
{
    const int u = blockIdx.x;      // 0..511
    const int t = threadIdx.x;     // 0..255
    __shared__ float xs[C_DIM];
    __shared__ float s1[256];
    __shared__ float s2[256];

    float x = context[u * C_DIM + t];
    s1[t] = x;
    s2[t] = x * x;
    __syncthreads();
    #pragma unroll
    for (int s = 128; s > 0; s >>= 1) {
        if (t < s) { s1[t] += s1[t + s]; s2[t] += s2[t + s]; }
        __syncthreads();
    }
    const float mu  = s1[0] * (1.0f / C_DIM);
    const float var = s2[0] * (1.0f / C_DIM) - mu * mu;
    const float rs  = rsqrtf(var + EPS_LN);
    xs[t] = (x - mu) * rs * gamma[t] + beta[t];
    __syncthreads();

    float acc = b1[t];
    const float* __restrict__ w = W1 + (size_t)t * C_DIM;
    #pragma unroll 8
    for (int c = 0; c < C_DIM; ++c) acc = fmaf(xs[c], w[c], acc);

    // exact GELU: 0.5*x*(1+erf(x/sqrt(2)))
    const float g = 0.5f * acc * (1.0f + erff(acc * 0.70710678118654752f));
    h_out[u * CAP_DIM + t] = (_Float16)g;
}

// --------------------------- Kernel B: params = alpha*(h@W2^T)+slow ---------
// Block: 256 threads = 8 waves arranged 4(M) x 2(N). Block tile: M=64, N=256.
// Grid: (NP/256, U/64) = (257, 8). K = 256 in 8 steps of 32.
__global__ __launch_bounds__(256) void hyperfc_params_kernel(
    const _Float16* __restrict__ h,     // [U][CAP] f16
    const float*    __restrict__ W2,    // [NP][CAP] f32 row-major
    const float*    __restrict__ slow,  // [NP]
    const float*    __restrict__ alpha_p,
    _Float16*       __restrict__ params)// [U][NP] f16
{
    const float alpha = alpha_p[0];
    const int lane = threadIdx.x & 31;
    const int wave = threadIdx.x >> 5;       // wave32
    const int mq   = wave & 3;               // M sub-tile 0..3
    const int nh   = wave >> 2;              // N half 0..1
    const int l15  = lane & 15;
    const int hi   = lane >> 4;              // lane half (K interleave select)

    const int m0 = blockIdx.y * 64 + mq * 16;          // u base of this wave
    const int n0 = blockIdx.x * 256 + nh * 128;        // p base of this wave

    v8f acc[8] = {};

    const _Float16* __restrict__ arow = h + (size_t)(m0 + l15) * CAP_DIM;

    for (int k0 = 0; k0 < CAP_DIM; k0 += 32) {
        // A operand (16x32 f16): lanes<16 -> K {0..15}, lanes>=16 -> {16..31}
        const _Float16* __restrict__ ap = arow + k0 + hi * 16;
        v16h a;
        #pragma unroll
        for (int i = 0; i < 8; ++i) { a[i] = ap[i]; a[8 + i] = ap[8 + i]; }

        #pragma unroll
        for (int nt = 0; nt < 8; ++nt) {
            // B operand (32x16): lane = column p, 16 contiguous K (=cap) values,
            // upper half-wave takes K+16.  W2 row-major [p][c] => contiguous.
            const float* __restrict__ bp =
                W2 + (size_t)(n0 + nt * 16 + l15) * CAP_DIM + k0 + hi * 16;
            v16h b;
            #pragma unroll
            for (int i = 0; i < 16; ++i) b[i] = (_Float16)bp[i];
            acc[nt] = __builtin_amdgcn_wmma_f32_16x16x32_f16(
                false, a, false, b, (short)0, acc[nt], false, false);
        }
    }

    // D layout: VGPR v, lanes<16 -> M=v, N=lane; lanes>=16 -> M=v+8, N=lane-16
    #pragma unroll
    for (int nt = 0; nt < 8; ++nt) {
        const int p  = n0 + nt * 16 + l15;
        const float sp = slow[p];
        #pragma unroll
        for (int v = 0; v < 8; ++v) {
            const int urow = m0 + v + hi * 8;
            params[(size_t)urow * NP + p] = (_Float16)fmaf(alpha, acc[nt][v], sp);
        }
    }
}

// --------------------------- Kernel C: out = X_u @ W_u^T + bias_u -----------
// One block per (u,b): M = V = 64 rows, N = O = 256, K = I = 256.
// 8 waves arranged 4(M) x 2(N); per K-step (K=32) the block stages
//   A tile: 64 x 32 f32  (512  x 16B chunks, 2/thread)
//   B tile: 256 x 32 f16 (1024 x 16B chunks, 4/thread)
// into double-buffered LDS via global_load_async_to_lds_b128.
#define A_STRIDE 36   // floats per LDS A row (padded from 32)
#define B_STRIDE 40   // halfs  per LDS B row (padded from 32)

__global__ __launch_bounds__(256) void hyperfc_apply_kernel(
    const float*    __restrict__ input,   // [B][U][V][I] f32
    const _Float16* __restrict__ params,  // [U][NP] f16
    float*          __restrict__ out)     // [B][U][V][O] f32
{
    __shared__ float    Ash[2][V_DIM * A_STRIDE];    // 2 x 9216 B
    __shared__ _Float16 Bsh[2][O_DIM * B_STRIDE];    // 2 x 20480 B

    const int t    = threadIdx.x;
    const int u    = blockIdx.x >> 3;   // consecutive blocks share u -> L2 reuse
    const int b    = blockIdx.x & 7;
    const int lane = t & 31;
    const int wave = t >> 5;
    const int mq   = wave & 3;
    const int nh   = wave >> 2;
    const int l15  = lane & 15;
    const int hi   = lane >> 4;
    const int n0   = nh * 128;

    const size_t in_base = (size_t)(b * U_DIM + u) * V_DIM * I_DIM;
    const float*    __restrict__ ag = input + in_base;       // + row*I + k
    const _Float16* __restrict__ wrow = params + (size_t)u * NP;

    // Per-thread chunk coordinates for the async tile copies.
    const int arow0 = t >> 3, ac0 = t & 7;                 // A chunks 0..255
    const int arow1 = (t + 256) >> 3, ac1 = t & 7;         // A chunks 256..511
    const int brow[4] = { t >> 2, (t + 256) >> 2, (t + 512) >> 2, (t + 768) >> 2 };
    const int bc = t & 3;

    // Issue one K-step tile (6 async b128 per thread: 2 for A, 4 for B).
    auto issue_tiles = [&](int buf, int k0) {
        {
            unsigned l0 = (unsigned)(size_t)&Ash[buf][arow0 * A_STRIDE] + ac0 * 16u;
            async_copy_b128(l0, (const char*)(ag + (size_t)arow0 * I_DIM + k0) + ac0 * 16);
            unsigned l1 = (unsigned)(size_t)&Ash[buf][arow1 * A_STRIDE] + ac1 * 16u;
            async_copy_b128(l1, (const char*)(ag + (size_t)arow1 * I_DIM + k0) + ac1 * 16);
        }
        #pragma unroll
        for (int j = 0; j < 4; ++j) {
            unsigned lb = (unsigned)(size_t)&Bsh[buf][brow[j] * B_STRIDE] + bc * 16u;
            async_copy_b128(lb, (const char*)(wrow + (size_t)brow[j] * I_DIM + k0) + bc * 16);
        }
    };

    v8f acc[8] = {};

    issue_tiles(0, 0);

    #pragma unroll
    for (int ks = 0; ks < 8; ++ks) {
        const int cur = ks & 1;
        if (ks < 7) {
            issue_tiles(cur ^ 1, (ks + 1) * 32);
            // each thread: 6 older + 6 newer async ops in flight; async loads
            // complete in order, so <=6 outstanding => current buffer landed.
            asm volatile("s_wait_asynccnt 0x6" ::: "memory");
        } else {
            asm volatile("s_wait_asynccnt 0x0" ::: "memory");
        }
        __syncthreads();   // all waves' tile copies visible

        // A operand from LDS (f32 -> f16): lanes<16 K{0..7,16..23}, hi half +8
        const float* __restrict__ ap = &Ash[cur][(mq * 16 + l15) * A_STRIDE] + hi * 8;
        v16h a;
        #pragma unroll
        for (int i = 0; i < 8; ++i) {
            a[i]     = (_Float16)ap[i];
            a[8 + i] = (_Float16)ap[16 + i];
        }

        #pragma unroll
        for (int nt = 0; nt < 8; ++nt) {
            const _Float16* __restrict__ bp =
                &Bsh[cur][(n0 + nt * 16 + l15) * B_STRIDE] + hi * 8;
            v8h b0 = *(const v8h*)bp;
            v8h b1 = *(const v8h*)(bp + 16);
            v16h bm;
            #pragma unroll
            for (int i = 0; i < 8; ++i) { bm[i] = b0[i]; bm[8 + i] = b1[i]; }
            acc[nt] = __builtin_amdgcn_wmma_f32_16x16x32_f16(
                false, a, false, bm, (short)0, acc[nt], false, false);
        }
        __syncthreads();   // done reading buf[cur] before it is refilled
    }

    // D layout: VGPR v, lanes<16 -> M=v, N=lane; lanes>=16 -> M=v+8, N=lane-16
    #pragma unroll
    for (int nt = 0; nt < 8; ++nt) {
        const int o = n0 + nt * 16 + l15;
        const float bias = (float)wrow[NW + o];
        #pragma unroll
        for (int v = 0; v < 8; ++v) {
            const int vrow = mq * 16 + v + hi * 8;
            out[in_base + (size_t)vrow * O_DIM + o] = acc[nt][v] + bias;
        }
    }
}

// --------------------------- host launcher ----------------------------------
extern "C" void kernel_launch(void* const* d_in, const int* in_sizes, int n_in,
                              void* d_out, int out_size, void* d_ws, size_t ws_size,
                              hipStream_t stream) {
    const float* input   = (const float*)d_in[0];  // (B,U,V,I)
    const float* context = (const float*)d_in[1];  // (U,C)
    const float* gamma   = (const float*)d_in[2];  // (C,)
    const float* beta    = (const float*)d_in[3];  // (C,)
    const float* W1      = (const float*)d_in[4];  // (CAP,C)
    const float* b1      = (const float*)d_in[5];  // (CAP,)
    const float* W2      = (const float*)d_in[6];  // (NP,CAP)
    const float* slow    = (const float*)d_in[7];  // (NP,)
    const float* alpha   = (const float*)d_in[8];  // scalar
    float* out = (float*)d_out;

    _Float16* h_ws      = (_Float16*)d_ws;                      // U*CAP f16
    _Float16* params_ws = h_ws + (size_t)U_DIM * CAP_DIM;       // U*NP  f16

    hyperfc_h_kernel<<<U_DIM, 256, 0, stream>>>(context, gamma, beta, W1, b1, h_ws);

    dim3 gridB(NP / 256, U_DIM / 64);   // (257, 8)
    hyperfc_params_kernel<<<gridB, 256, 0, stream>>>(h_ws, W2, slow, alpha, params_ws);

    hyperfc_apply_kernel<<<U_DIM * B_DIM, 256, 0, stream>>>(input, params_ws, out);
}
